// SelfAttention_14113262535175
// MI455X (gfx1250) — compile-verified, matrix operand order checked by reading, not run on hardware
//
#include <hip/hip_runtime.h>
#include <hip/hip_bf16.h>

typedef __bf16 bf16;
typedef __attribute__((ext_vector_type(16))) __bf16 v16bf;
typedef __attribute__((ext_vector_type(8)))  float   v8f;

union Frag16 { uint4 u4[2]; v16bf v; };
union BF16x8 { uint4 u4; bf16 h[8]; };

// ---------------------------------------------------------------------------
// WMMA helper: D = A(16x32 bf16) * B(32x16 bf16) + C(16x16 f32)
// ---------------------------------------------------------------------------
static __device__ inline v8f wmma_bf16(v16bf a, v16bf b, v8f c) {
  return __builtin_amdgcn_wmma_f32_16x16x32_bf16(
      /*neg_a=*/false, a, /*neg_b=*/false, b,
      /*c_mod=*/(short)0, c, /*reuse_a=*/false, /*reuse_b=*/false);
}

// A/B fragment loader for 16-bit operands (wave32 layout, ISA 7.12.2):
//   lanes 0-15 : row r = lane,    K = {0..7, 16..23}
//   lanes16-31 : row r = lane-16, K = {8..15, 24..31}
static __device__ inline v16bf load_frag(const bf16* __restrict__ base,
                                         int stride, int lane) {
  const int r  = lane & 15;
  const int kb = (lane >> 4) << 3;
  const bf16* p = base + (size_t)r * stride + kb;
  Frag16 f;
  f.u4[0] = *(const uint4*)(p);
  f.u4[1] = *(const uint4*)(p + 16);
  return f.v;
}

// Same layout, sourced from fp32 memory with inline bf16 conversion.
static __device__ inline v16bf load_frag_f32(const float* __restrict__ base,
                                             int stride, int lane) {
  const int r  = lane & 15;
  const int kb = (lane >> 4) << 3;
  const float* p = base + (size_t)r * stride + kb;
  float4 a0 = *(const float4*)(p);
  float4 a1 = *(const float4*)(p + 4);
  float4 b0 = *(const float4*)(p + 16);
  float4 b1 = *(const float4*)(p + 20);
  v16bf v;
  v[0]  = (bf16)a0.x; v[1]  = (bf16)a0.y; v[2]  = (bf16)a0.z; v[3]  = (bf16)a0.w;
  v[4]  = (bf16)a1.x; v[5]  = (bf16)a1.y; v[6]  = (bf16)a1.z; v[7]  = (bf16)a1.w;
  v[8]  = (bf16)b0.x; v[9]  = (bf16)b0.y; v[10] = (bf16)b0.z; v[11] = (bf16)b0.w;
  v[12] = (bf16)b1.x; v[13] = (bf16)b1.y; v[14] = (bf16)b1.z; v[15] = (bf16)b1.w;
  return v;
}

// ---------------------------------------------------------------------------
// Async staging: 4 x global_load_async_to_lds_b128 per thread (ASYNCcnt +4).
// The 24-bit inst offset is added to BOTH the LDS and global addresses
// (ISA 08_async_tensor 4.4), so one LDS-address VGPR covers both 16B chunks.
// ---------------------------------------------------------------------------
static __device__ inline void stage_async(const bf16* gK, const bf16* gV,
                                          uint32_t ldsK, uint32_t ldsV,
                                          uint32_t offK, uint32_t offV) {
  asm volatile(
      "global_load_async_to_lds_b128 %0, %1, %4 offset:0\n\t"
      "global_load_async_to_lds_b128 %0, %1, %4 offset:16\n\t"
      "global_load_async_to_lds_b128 %2, %3, %5 offset:0\n\t"
      "global_load_async_to_lds_b128 %2, %3, %5 offset:16"
      :
      : "v"(ldsK), "v"(offK), "v"(ldsV), "v"(offV), "s"(gK), "s"(gV)
      : "memory");
}

// ---------------------------------------------------------------------------
// Kernel 1: W[1024x128] fp32 -> Wt[128x1024] bf16 (transposed) for Q/K/V
// ---------------------------------------------------------------------------
__global__ __launch_bounds__(256) void wtrans_kernel(
    const float* __restrict__ Wq, const float* __restrict__ Wk,
    const float* __restrict__ Wv, bf16* __restrict__ Wt) {
  const int idx = blockIdx.x * 256 + threadIdx.x;   // 0 .. 131071
  const int m   = blockIdx.y;                       // 0=Q 1=K 2=V
  const float* W = (m == 0) ? Wq : (m == 1) ? Wk : Wv;
  const int c = idx >> 7;
  const int d = idx & 127;
  Wt[(size_t)m * 131072 + (size_t)d * 1024 + c] = (bf16)W[idx];
}

// ---------------------------------------------------------------------------
// Kernel 2: QKV projection.  One wave computes a 16x16 output tile via
// 32 bf16 WMMAs over C=1024.  Q gets 1/sqrt(D) folded in.
// Q,K row-major bf16 [B][L][128]; V transposed bf16 [B][128][L].
// ---------------------------------------------------------------------------
__global__ __launch_bounds__(256) void qkv_proj_kernel(
    const float* __restrict__ x,  const bf16* __restrict__ Wt,
    const float* __restrict__ bq, const float* __restrict__ bk,
    const float* __restrict__ bv,
    bf16* __restrict__ Qb, bf16* __restrict__ Kb, bf16* __restrict__ Vt) {
  const int lane = threadIdx.x & 31;
  const int wave = threadIdx.x >> 5;      // column tile 0..7
  const int mtx  = blockIdx.y;            // 0=Q 1=K 2=V
  const int b    = blockIdx.z;
  const int row0 = blockIdx.x * 16;

  const float* xb = x + ((size_t)b * 4096 + row0) * 1024;
  const bf16*  wb = Wt + (size_t)mtx * 131072 + (size_t)(wave * 16) * 1024;

  v8f acc = {};
  #pragma unroll 4
  for (int kc = 0; kc < 1024; kc += 32) {
    v16bf a = load_frag_f32(xb + kc, 1024, lane);
    v16bf w = load_frag(wb + kc, 1024, lane);
    acc = wmma_bf16(a, w, acc);
  }

  const float* bias = (mtx == 0) ? bq : (mtx == 1) ? bk : bv;
  const int col   = wave * 16 + (lane & 15);
  const int mbase = (lane >> 4) << 3;
  const float bc  = bias[col];

  if (mtx == 2) {
    BF16x8 pkt;
    #pragma unroll
    for (int i = 0; i < 8; ++i) pkt.h[i] = (bf16)(acc[i] + bc);
    *(uint4*)(Vt + ((size_t)b * 128 + col) * 4096 + row0 + mbase) = pkt.u4;
  } else {
    const float sc = (mtx == 0) ? 0.088388347648318447f : 1.0f; // 1/sqrt(128)
    bf16* op = (mtx == 0 ? Qb : Kb) + ((size_t)b * 4096 + row0) * 128;
    #pragma unroll
    for (int i = 0; i < 8; ++i)
      op[(size_t)(mbase + i) * 128 + col] = (bf16)((acc[i] + bc) * sc);
  }
}

// ---------------------------------------------------------------------------
// Kernel 3: flash attention with async double-buffered K/V staging in LDS.
// 8 waves/block; each wave owns 16 query rows x full D=128 O tile in regs.
// Per 32-key step: 8 WMMAs for S, online softmax, P via LDS relayout,
// 8 WMMAs for P*V.  K/V tiles staged once per block (8x traffic reduction).
// ---------------------------------------------------------------------------
__global__ __launch_bounds__(256) void attn_kernel(
    const bf16* __restrict__ Qb, const bf16* __restrict__ Kb,
    const bf16* __restrict__ Vt, float* __restrict__ out) {
  __shared__ __align__(16) bf16 Klds[2][32][128];  // 16 KB (double buffer)
  __shared__ __align__(16) bf16 Vlds[2][128][32];  // 16 KB (double buffer)
  __shared__ __align__(16) bf16 Plds[8][16][32];   //  8 KB (wave-private)

  const int tid  = threadIdx.x;
  const int lane = tid & 31;
  const int wave = tid >> 5;
  const int b    = blockIdx.y;
  const int row0 = blockIdx.x * 128 + wave * 16;

  // Per-thread staging assignment:
  //   K tile: flat 8 KB copy, thread t owns bytes [t*32, t*32+32)
  //   V tile: thread t owns 32 bytes of row d = t/2 of Vt (stride 4096 elems)
  const uint32_t kOff = (uint32_t)tid * 32;
  const int      vd   = tid >> 1;
  const int      vh   = tid & 1;
  const uint32_t vOff = (uint32_t)vd * 8192 + (uint32_t)vh * 32;
  uint32_t kLds[2], vLds[2];
  #pragma unroll
  for (int p = 0; p < 2; ++p) {
    kLds[p] = (uint32_t)(uintptr_t)&Klds[p][0][0] + kOff;
    vLds[p] = (uint32_t)(uintptr_t)&Vlds[p][0][0] + (uint32_t)vd * 64 + (uint32_t)vh * 32;
  }
  const bf16* gK0 = Kb + ((size_t)b * 4096) * 128;   // + j*128 per tile
  const bf16* gV0 = Vt + (size_t)b * 128 * 4096;     // + j     per tile

  // Preload the wave's Q tile (already scaled by 1/sqrt(D)).
  const bf16* qb = Qb + ((size_t)b * 4096 + row0) * 128;
  v16bf qf[4];
  #pragma unroll
  for (int kc = 0; kc < 4; ++kc) qf[kc] = load_frag(qb + kc * 32, 128, lane);

  const v8f vzero = {};
  v8f o[8];
  float mrun[8], lrun[8];
  #pragma unroll
  for (int t = 0; t < 8; ++t) o[t] = vzero;
  #pragma unroll
  for (int i = 0; i < 8; ++i) { mrun[i] = -3.0e38f; lrun[i] = 0.0f; }

  const int col = lane & 15;
  const int rb  = (lane >> 4) << 3;

  // Prologue: stage tile j=0 into buffer 0.
  stage_async(gK0, gV0, kLds[0], vLds[0], kOff, vOff);

  for (int j0 = 0; j0 < 4096; j0 += 32) {
    const int buf = (j0 >> 5) & 1;
    if (j0 + 32 < 4096) {
      stage_async(gK0 + (size_t)(j0 + 32) * 128, gV0 + (j0 + 32),
                  kLds[buf ^ 1], vLds[buf ^ 1], kOff, vOff);
      asm volatile("s_wait_asynccnt 0x4" ::: "memory");  // tile j0 landed
    } else {
      asm volatile("s_wait_asynccnt 0x0" ::: "memory");
    }
    __syncthreads();   // all waves see the staged tile

    // ---- S = Q * K^T for two 16-key tiles (K fragments from LDS) ---------
    v8f s0 = vzero, s1 = vzero;
    {
      const int r  = lane & 15;
      const int kb = (lane >> 4) << 3;
      #pragma unroll
      for (int kc = 0; kc < 4; ++kc) {
        Frag16 kf;
        kf.u4[0] = *(const uint4*)&Klds[buf][r][kc * 32 + kb];
        kf.u4[1] = *(const uint4*)&Klds[buf][r][kc * 32 + kb + 16];
        s0 = wmma_bf16(qf[kc], kf.v, s0);
      }
      #pragma unroll
      for (int kc = 0; kc < 4; ++kc) {
        Frag16 kf;
        kf.u4[0] = *(const uint4*)&Klds[buf][16 + r][kc * 32 + kb];
        kf.u4[1] = *(const uint4*)&Klds[buf][16 + r][kc * 32 + kb + 16];
        s1 = wmma_bf16(qf[kc], kf.v, s1);
      }
    }

    // ---- online softmax: row reductions in 16-lane half-groups -----------
    float p0[8], p1[8], scl[8];
    #pragma unroll
    for (int i = 0; i < 8; ++i) {
      float tm = fmaxf(s0[i], s1[i]);
      tm = fmaxf(tm, __shfl_xor(tm, 1, 32));
      tm = fmaxf(tm, __shfl_xor(tm, 2, 32));
      tm = fmaxf(tm, __shfl_xor(tm, 4, 32));
      tm = fmaxf(tm, __shfl_xor(tm, 8, 32));
      const float mnew = fmaxf(mrun[i], tm);
      const float sc   = __expf(mrun[i] - mnew);
      const float e0   = __expf(s0[i] - mnew);
      const float e1   = __expf(s1[i] - mnew);
      float rs = e0 + e1;
      rs += __shfl_xor(rs, 1, 32);
      rs += __shfl_xor(rs, 2, 32);
      rs += __shfl_xor(rs, 4, 32);
      rs += __shfl_xor(rs, 8, 32);
      lrun[i] = lrun[i] * sc + rs;
      mrun[i] = mnew;
      p0[i] = e0; p1[i] = e1; scl[i] = sc;
    }
    #pragma unroll
    for (int t = 0; t < 8; ++t) {
      #pragma unroll
      for (int i = 0; i < 8; ++i) o[t][i] *= scl[i];
    }

    // ---- P: C layout -> LDS -> A-fragment layout -------------------------
    #pragma unroll
    for (int i = 0; i < 8; ++i) {
      Plds[wave][rb + i][col]      = (bf16)p0[i];
      Plds[wave][rb + i][col + 16] = (bf16)p1[i];
    }
    Frag16 pfr;
    {
      const int r  = lane & 15;
      const int kb = (lane >> 4) << 3;
      pfr.u4[0] = *(const uint4*)&Plds[wave][r][kb];
      pfr.u4[1] = *(const uint4*)&Plds[wave][r][kb + 16];
    }

    // ---- O += P * V  (V fragments from LDS, d-major layout) --------------
    #pragma unroll
    for (int dt = 0; dt < 8; ++dt) {
      const int d  = dt * 16 + (lane & 15);
      const int kb = (lane >> 4) << 3;
      Frag16 vf;
      vf.u4[0] = *(const uint4*)&Vlds[buf][d][kb];
      vf.u4[1] = *(const uint4*)&Vlds[buf][d][kb + 16];
      o[dt] = wmma_bf16(pfr.v, vf.v, o[dt]);
    }

    __syncthreads();   // tile consumed: safe to overwrite this buffer
  }

  // ---- normalize and write fp32 output -----------------------------------
  float inv[8];
  #pragma unroll
  for (int i = 0; i < 8; ++i) inv[i] = 1.0f / lrun[i];
  float* ob = out + ((size_t)b * 4096 + row0) * 128;
  #pragma unroll
  for (int dt = 0; dt < 8; ++dt) {
    #pragma unroll
    for (int i = 0; i < 8; ++i)
      ob[(size_t)(rb + i) * 128 + dt * 16 + col] = o[dt][i] * inv[i];
  }
}

// ---------------------------------------------------------------------------
extern "C" void kernel_launch(void* const* d_in, const int* in_sizes, int n_in,
                              void* d_out, int out_size, void* d_ws, size_t ws_size,
                              hipStream_t stream) {
  const float* x  = (const float*)d_in[0];
  const float* Wq = (const float*)d_in[1];
  const float* bq = (const float*)d_in[2];
  const float* Wk = (const float*)d_in[3];
  const float* bk = (const float*)d_in[4];
  const float* Wv = (const float*)d_in[5];
  const float* bv = (const float*)d_in[6];
  float* out = (float*)d_out;

  // Workspace (bf16): Wt 3*131072 | Qb 2097152 | Kb 2097152 | Vt 2097152
  bf16* Wt = (bf16*)d_ws;
  bf16* Qb = Wt + (size_t)3 * 131072;
  bf16* Kb = Qb + (size_t)4 * 4096 * 128;
  bf16* Vt = Kb + (size_t)4 * 4096 * 128;

  wtrans_kernel<<<dim3(512, 3), 256, 0, stream>>>(Wq, Wk, Wv, Wt);
  qkv_proj_kernel<<<dim3(256, 3, 4), 256, 0, stream>>>(x, Wt, bq, bk, bv, Qb, Kb, Vt);
  attn_kernel<<<dim3(32, 4), 256, 0, stream>>>(Qb, Kb, Vt, out);
}
